// TrustBiasedGAT_87514253623511
// MI455X (gfx1250) — matrix operand order repaired and checked by reading, not compiled, not run on hardware
//
#include <hip/hip_runtime.h>

// Trust-biased GAT forward for MI455X (gfx1250, wave32).
// One block per batch row; K/V projections via v_wmma_f32_16x16x32_bf16.
// Neighbor tiles are streamed HBM->LDS with GLOBAL_LOAD_ASYNC_TO_LDS_B128
// (ASYNCcnt) so global traffic overlaps the WMMA + attention phases.

#define BATCH   1024
#define NBRS    512
#define DIM     128
#define HEADS   4
#define HDIM    32
#define CHUNK   64
#define NCHUNK  (NBRS / CHUNK)      // 8
#define WSTRIDE 136                 // bf16 elems/row (16B pad -> 4-bank skew)
#define KSTRIDE 132                 // f32 elems/row  (16B pad -> 4-bank skew)
#define RSTRIDE 132                 // f32 elems/row for raw staging buffer

typedef __bf16 v16bf __attribute__((ext_vector_type(16)));
typedef float  v8f   __attribute__((ext_vector_type(8)));

union Frag { uint4 q[2]; v16bf v; };

__device__ __forceinline__ unsigned short f2bf(float f) {
  unsigned u = __float_as_uint(f);
  u += 0x7FFFu + ((u >> 16) & 1u);          // round-to-nearest-even
  return (unsigned short)(u >> 16);
}
__device__ __forceinline__ unsigned pack2(float a, float b) {
  return (unsigned)f2bf(a) | ((unsigned)f2bf(b) << 16);
}

__device__ __forceinline__ void wait_async0() {
#if __has_builtin(__builtin_amdgcn_s_wait_asynccnt)
  __builtin_amdgcn_s_wait_asynccnt(0);
#else
  asm volatile("s_wait_asynccnt 0" ::: "memory");
#endif
}

// Issue one 64x128 fp32 chunk: HBM -> LDS raw staging, 8x b128 per thread.
__device__ __forceinline__ void async_stage_chunk(const float* gbase_uniform,
                                                  unsigned ldsBase, int t) {
  unsigned long long sbase = (unsigned long long)(uintptr_t)gbase_uniform;
  #pragma unroll
  for (int i = 0; i < 8; ++i) {
    int f4  = t + i * 256;                     // 0..2047 (float4 index)
    int row = f4 >> 5;
    int col = f4 & 31;
    unsigned voff  = (unsigned)f4 * 16u;                               // global byte offset
    unsigned laddr = ldsBase + (unsigned)row * (RSTRIDE * 4u)
                             + (unsigned)col * 16u;                    // LDS byte addr
    asm volatile("global_load_async_to_lds_b128 %0, %1, %2"
                 :: "v"(laddr), "v"(voff), "s"(sbase)
                 : "memory");
  }
}

// LDS carve (bytes):
//   sWkT 34816 | sWvT 34816 | sX 17408 | sK 33792 | sV 33792 | misc 11808 | sXraw 33792
#define SMEM_BYTES 200224

__global__ __launch_bounds__(256)
void TrustBiasedGAT_kernel(const float* __restrict__ ego,
                           const float* __restrict__ nbr,
                           const float* __restrict__ trust,
                           const unsigned char* __restrict__ mask,
                           const float* __restrict__ Wq, const float* __restrict__ bq,
                           const float* __restrict__ Wk, const float* __restrict__ bk,
                           const float* __restrict__ Wv, const float* __restrict__ bv,
                           const float* __restrict__ Wo, const float* __restrict__ bo,
                           const float* __restrict__ gainp,
                           float* __restrict__ out, float* __restrict__ attn)
{
  extern __shared__ unsigned char smem[];
  unsigned short* sWkT = (unsigned short*)smem;            // [128][136] bf16
  unsigned short* sWvT = sWkT + DIM * WSTRIDE;             // [128][136] bf16
  unsigned short* sX   = sWvT + DIM * WSTRIDE;             // [64][136]  bf16
  float* sK    = (float*)(sX + CHUNK * WSTRIDE);           // [64][132]  f32
  float* sV    = sK + CHUNK * KSTRIDE;                     // [64][132]  f32
  float* sQ    = sV + CHUNK * KSTRIDE;                     // [128]
  float* sBk   = sQ   + DIM;                               // [128]
  float* sBv   = sBk  + DIM;                               // [128]
  float* sCtx  = sBv  + DIM;                               // [128]
  float* sEgo  = sCtx + DIM;                               // [128]
  float* sLog  = sEgo + DIM;                               // [4][512]
  float* sP    = sLog + HEADS * NBRS;                      // [4][64]
  float* sMF   = sP   + HEADS * CHUNK;                     // [4]
  float* sLF   = sMF  + HEADS;                             // [4]
  float* sXraw = sLF  + HEADS;                             // [64][132] f32 staging

  const int b    = blockIdx.x;
  const int t    = threadIdx.x;
  const int w    = t >> 5;
  const int lane = t & 31;
  const float gain = gainp[0];

  const float* nbrB = nbr + (size_t)b * NBRS * DIM;
  const unsigned sXrawOff = (unsigned)(uintptr_t)(void*)sXraw;  // LDS byte offset

  // Kick off chunk 0 async stream immediately; it flies during weight staging.
  async_stage_chunk(nbrB, sXrawOff, t);

  // ---- Stage transposed bf16 weights + biases + ego ----
  #pragma unroll 4
  for (int i = 0; i < 64; ++i) {
    int idx = t + i * 256;            // e*128 + h, coalesced global reads
    int e = idx >> 7, h = idx & 127;
    sWkT[h * WSTRIDE + e] = f2bf(Wk[idx]);
    sWvT[h * WSTRIDE + e] = f2bf(Wv[idx]);
  }
  if (t < DIM) {
    sBk[t]  = bk[t];
    sBv[t]  = bv[t];
    sEgo[t] = ego[(size_t)b * DIM + t];
  }
  __syncthreads();

  // ---- q = ego @ Wq + bq (tiny matvec, fp32 VALU) ----
  if (t < DIM) {
    float acc = bq[t];
    #pragma unroll 4
    for (int e = 0; e < DIM; ++e) acc = fmaf(sEgo[e], Wq[e * DIM + t], acc);
    sQ[t] = acc;
  }

  // ---- per-wave WMMA role: waves 0-3 -> K tiles, waves 4-7 -> V tiles ----
  const bool isK = (w < 4);
  const int  mt  = isK ? w : (w - 4);
  const unsigned short* sWT   = isK ? sWkT : sWvT;
  float*                sOut  = isK ? sK   : sV;
  const float*          sBias = isK ? sBk  : sBv;
  const int rowA  = mt * 16 + (lane & 15);
  const int halfA = lane >> 4;

  // online-softmax state (valid for t<128: head = w, ctx dim = lane)
  float mrun = -1e30f, lrun = 0.f, ctx = 0.f;

  for (int ch = 0; ch < NCHUNK; ++ch) {
    wait_async0();    // my wave's async b128s for chunk ch have landed in sXraw
    __syncthreads();  // (A) all waves' async data visible; prior readers done

    // ---- phase 1: convert staged fp32 -> bf16 WMMA layout (LDS->LDS) ----
    #pragma unroll
    for (int i = 0; i < 8; ++i) {
      int f4 = t + i * 256;                    // 0..2047
      int row = f4 >> 5;
      int col = (f4 & 31) * 4;
      float4 vv = *(const float4*)&sXraw[row * RSTRIDE + col];
      *(uint2*)&sX[row * WSTRIDE + col] =
          make_uint2(pack2(vv.x, vv.y), pack2(vv.z, vv.w));
    }
    __syncthreads();  // (B) sX ready; all sXraw reads complete

    // stream next chunk from HBM while WMMA + attention run
    if (ch + 1 < NCHUNK)
      async_stage_chunk(nbrB + (size_t)(ch + 1) * CHUNK * DIM, sXrawOff, t);

    // ---- phase 2: K/V = X @ W via v_wmma_f32_16x16x32_bf16 ----
    v8f acc[8];
    const v8f vzero = {0.f, 0.f, 0.f, 0.f, 0.f, 0.f, 0.f, 0.f};
    #pragma unroll
    for (int nt = 0; nt < 8; ++nt) acc[nt] = vzero;

    #pragma unroll
    for (int kb = 0; kb < DIM; kb += 32) {
      Frag A;   // A 16x32 bf16: contiguous 32B of X row per lane
      const uint4* pa = (const uint4*)&sX[rowA * WSTRIDE + kb + halfA * 16];
      A.q[0] = pa[0];
      A.q[1] = pa[1];
      #pragma unroll
      for (int nt = 0; nt < 8; ++nt) {
        Frag Bf;  // B 32x16 bf16: contiguous 32B of W^T row per lane
        const uint4* pb = (const uint4*)
            &sWT[(nt * 16 + (lane & 15)) * WSTRIDE + kb + halfA * 16];
        Bf.q[0] = pb[0];
        Bf.q[1] = pb[1];
        acc[nt] = __builtin_amdgcn_wmma_f32_16x16x32_bf16(
            false, A.v, false, Bf.v, (short)0, acc[nt], false, false);
      }
    }
    // scatter C tiles (+bias) to fp32 LDS; C layout: VGPR i -> M=i+8*half, N=lane%16
    #pragma unroll
    for (int nt = 0; nt < 8; ++nt) {
      #pragma unroll
      for (int i = 0; i < 8; ++i) {
        int row = mt * 16 + i + 8 * halfA;
        int col = nt * 16 + (lane & 15);
        sOut[row * KSTRIDE + col] = acc[nt][i] + sBias[col];
      }
    }
    __syncthreads();  // (C)

    // ---- phase 3: logits (fp32 VALU) ----
    {
      int h = t >> 6, nl = t & 63, n = ch * CHUNK + nl;
      const float* kr = &sK[nl * KSTRIDE + h * HDIM];
      const float* qr = &sQ[h * HDIM];
      float dot = 0.f;
      #pragma unroll
      for (int d = 0; d < HDIM; ++d) dot = fmaf(qr[d], kr[d], dot);
      float lg = dot * 0.17677669529663687f            // 1/sqrt(32)
               + trust[(size_t)b * NBRS + n] * gain;
      if (!mask[(size_t)b * NBRS + n]) lg = -1e9f;
      sLog[h * NBRS + n] = lg;
    }
    __syncthreads();  // (D)

    // ---- phase 4a: per-head online softmax update (waves 0-3) ----
    float scale = 0.f, lsum = 0.f, mnew = mrun;
    if (t < 128) {
      int h = w, j = lane;
      float l0 = sLog[h * NBRS + ch * CHUNK + j];
      float l1 = sLog[h * NBRS + ch * CHUNK + 32 + j];
      float mloc = fmaxf(l0, l1);
      #pragma unroll
      for (int off = 16; off; off >>= 1)
        mloc = fmaxf(mloc, __shfl_xor(mloc, off, 32));
      mnew  = fmaxf(mrun, mloc);
      scale = __expf(mrun - mnew);
      float p0 = __expf(l0 - mnew);
      float p1 = __expf(l1 - mnew);
      sP[h * CHUNK + j]      = p0;
      sP[h * CHUNK + 32 + j] = p1;
      lsum = p0 + p1;
      #pragma unroll
      for (int off = 16; off; off >>= 1)
        lsum += __shfl_xor(lsum, off, 32);
    }
    __syncthreads();  // (E)

    // ---- phase 4b: context accumulation, ctx dim = lane ----
    if (t < 128) {
      int h = w, j = lane;
      float c = ctx * scale;
      const float* vp = &sV[h * HDIM + j];
      const float* pp = &sP[h * CHUNK];
      #pragma unroll 8
      for (int nl = 0; nl < CHUNK; ++nl)
        c = fmaf(pp[nl], vp[nl * KSTRIDE], c);
      ctx  = c;
      lrun = lrun * scale + lsum;
      mrun = mnew;
    }
  }

  // ---- finalize ----
  if (t < 128) {
    int h = w, j = lane;
    float inv = 1.f / lrun;           // lrun >= 1 (max elem contributes exp(0))
    sCtx[h * HDIM + j] = ctx * inv;
    if (j == 0) { sMF[h] = mrun; sLF[h] = lrun; }
  }
  __syncthreads();

  // attn output (B, HEADS, NBRS): masked exp underflows to exactly 0 -> denom==1
  float* attnB = attn + (size_t)b * HEADS * NBRS;
  #pragma unroll
  for (int i = 0; i < 8; ++i) {
    int idx = t + i * 256;            // = h*512 + n
    int h = idx >> 9, n = idx & 511;
    float a = __expf(sLog[idx] - sMF[h]) / sLF[h];
    if (!mask[(size_t)b * NBRS + n]) a = 0.f;
    attnB[idx] = a;
  }

  // out = ctx @ Wo + bo
  if (t < 128) {
    float o = bo[t];
    #pragma unroll 4
    for (int e = 0; e < DIM; ++e) o = fmaf(sCtx[e], Wo[e * DIM + t], o);
    out[(size_t)b * DIM + t] = o;
  }
}

extern "C" void kernel_launch(void* const* d_in, const int* in_sizes, int n_in,
                              void* d_out, int out_size, void* d_ws, size_t ws_size,
                              hipStream_t stream) {
  (void)in_sizes; (void)n_in; (void)out_size; (void)d_ws; (void)ws_size;
  const float*         ego   = (const float*)d_in[0];
  const float*         nbr   = (const float*)d_in[1];
  const float*         trust = (const float*)d_in[2];
  const unsigned char* mask  = (const unsigned char*)d_in[3];
  const float* Wq = (const float*)d_in[4];
  const float* bq = (const float*)d_in[5];
  const float* Wk = (const float*)d_in[6];
  const float* bk = (const float*)d_in[7];
  const float* Wv = (const float*)d_in[8];
  const float* bv = (const float*)d_in[9];
  const float* Wo = (const float*)d_in[10];
  const float* bo = (const float*)d_in[11];
  const float* gain = (const float*)d_in[12];

  float* out  = (float*)d_out;                     // (B, 128)
  float* attn = out + (size_t)BATCH * DIM;         // (B, 4, 512)

  hipFuncSetAttribute((const void*)TrustBiasedGAT_kernel,
                      hipFuncAttributeMaxDynamicSharedMemorySize, SMEM_BYTES);
  TrustBiasedGAT_kernel<<<BATCH, 256, SMEM_BYTES, stream>>>(
      ego, nbr, trust, mask, Wq, bq, Wk, bk, Wv, bv, Wo, bo, gain, out, attn);
}
